// Generator_18262200943059
// MI455X (gfx1250) — matrix operand order repaired and checked by reading, not compiled, not run on hardware
//
#include <hip/hip_runtime.h>

// ---------------------------------------------------------------------------
// StyleGAN2-like generator for gfx1250 (MI455X).
// Core convs run as implicit-GEMM on v_wmma_f32_16x16x32_f16 (wave32 WMMA),
// activations f16 NHWC, modulated weights packed f16 [b][tap][co][ci].
// Invalid im2col columns (halo / inserted zeros of the transposed conv) are
// redirected to a 1KB zero page so the WMMA inner loop has NO exec-mask
// branching: just unconditional global_load_b128 pairs feeding v_wmma.
// Workspace budget (~190 MB assumed available in d_ws):
//   [0KB]      w latent  (4x512 f32)
//   [16KB]     style s   (4x512 f32)
//   [32KB]     rgb wgt   (4x3x512 f32)
//   [48KB]     zero page (512 f16)
//   [1MB]      packed conv weights (<= 4*9*512*512 f16 = 18.9MB)
//   [24MB]     act buffer A (<= 68MB)
//   [100MB]    act buffer B (<= 68MB)
//   [176MB]    skip A (f32 NCHW, <= 0.8MB mid-res)
//   [184MB]    skip B
// ---------------------------------------------------------------------------

typedef __attribute__((ext_vector_type(16))) _Float16 v16h;
typedef __attribute__((ext_vector_type(8)))  _Float16 v8h;
typedef __attribute__((ext_vector_type(8)))  float    v8f;

#define SQRT2F 1.4142135623730951f
#define INV_SQRT_WDIM 0.044194173824159216f  // 1/sqrt(512)

__device__ __forceinline__ float lrelu2(float v) {
    return (v > 0.0f ? v : 0.2f * v) * SQRT2F;
}

// --------------------------- zero page -------------------------------------
__global__ void zero_page_kernel(_Float16* __restrict__ z)
{
    int idx = blockIdx.x * blockDim.x + threadIdx.x;
    if (idx < 512) z[idx] = (_Float16)0.0f;
}

// --------------------------- mapping network -------------------------------
// z[4,512] -> PixelNorm -> 8x (EqualLinear 512x512, lr_mul 0.01, lrelu) -> w
__global__ __launch_bounds__(512)
void mapping_kernel(const float* __restrict__ z, const float* __restrict__ mw,
                    const float* __restrict__ mb, float* __restrict__ wlat)
{
    __shared__ float xs[4][512];
    __shared__ float scl[4];
    const int t = threadIdx.x;
    for (int b = 0; b < 4; ++b) xs[b][t] = z[b * 512 + t];
    __syncthreads();
    if (t < 4) {
        float s = 0.f;
        for (int k = 0; k < 512; ++k) s += xs[t][k] * xs[t][k];
        scl[t] = rsqrtf(s * (1.0f / 512.0f) + 1e-8f);
    }
    __syncthreads();
    float mysc0 = scl[0], mysc1 = scl[1], mysc2 = scl[2], mysc3 = scl[3];
    xs[0][t] *= mysc0; xs[1][t] *= mysc1; xs[2][t] *= mysc2; xs[3][t] *= mysc3;
    __syncthreads();
    const float sc = 0.01f * INV_SQRT_WDIM;
    for (int l = 0; l < 8; ++l) {
        float acc[4];
        const float bias = mb[l * 512 + t] * 0.01f;
        #pragma unroll
        for (int b = 0; b < 4; ++b) acc[b] = bias;
        const float* wr = mw + ((size_t)l * 512 + t) * 512;
        for (int k = 0; k < 512; ++k) {
            const float wv = wr[k] * sc;
            #pragma unroll
            for (int b = 0; b < 4; ++b) acc[b] += xs[b][k] * wv;
        }
        __syncthreads();
        #pragma unroll
        for (int b = 0; b < 4; ++b) xs[b][t] = lrelu2(acc[b]);
        __syncthreads();
    }
    for (int b = 0; b < 4; ++b) wlat[b * 512 + t] = xs[b][t];
}

// --------------------------- style modulation ------------------------------
// s[b][ci] = dot(w[b], mod_w[ci]) / sqrt(512) + mod_b[ci]
__global__ void style_kernel(const float* __restrict__ wlat,
                             const float* __restrict__ mod_w,
                             const float* __restrict__ mod_b,
                             float* __restrict__ s, int Ci)
{
    int idx = blockIdx.x * blockDim.x + threadIdx.x;
    if (idx >= 4 * Ci) return;
    int ci = idx % Ci, b = idx / Ci;
    const float* wr = wlat + b * 512;
    const float* mr = mod_w + (size_t)ci * 512;
    float acc = 0.f;
    for (int k = 0; k < 512; ++k) acc += wr[k] * mr[k];
    s[idx] = acc * INV_SQRT_WDIM + mod_b[ci];
}

// Modulate + demodulate a 3x3 conv, pack f16 as [b][tap][co][ci] (tap
// spatially flipped for transposed-conv layers). One block per (co,b).
__global__ __launch_bounds__(256)
void modpack_kernel(const float* __restrict__ w0, const float* __restrict__ s,
                    _Float16* __restrict__ wgt, int Ci, int Co, int flip)
{
    const int co = blockIdx.x, b = blockIdx.y;
    const int n = Ci * 9;
    const float scale = rsqrtf((float)(Ci * 9));
    __shared__ float red[256];
    const float* wrow = w0 + (size_t)co * n;   // [co][ci][3][3]
    const float* sb = s + (size_t)b * Ci;
    float part = 0.f;
    for (int i = threadIdx.x; i < n; i += 256) {
        int ci = i / 9;
        float v = scale * wrow[i] * sb[ci];
        part += v * v;
    }
    red[threadIdx.x] = part;
    __syncthreads();
    for (int st = 128; st > 0; st >>= 1) {
        if (threadIdx.x < st) red[threadIdx.x] += red[threadIdx.x + st];
        __syncthreads();
    }
    const float dm = rsqrtf(red[0] + 1e-8f);
    for (int i = threadIdx.x; i < n; i += 256) {
        int ci = i / 9, tap = i - ci * 9;
        int tp = flip ? (8 - tap) : tap;
        float v = scale * wrow[i] * sb[ci] * dm;
        wgt[(((size_t)b * 9 + tp) * Co + co) * Ci + ci] = (_Float16)v;
    }
}

// toRGB 1x1 modulated weights, no demod: rgbw[b][c][ci] f32
__global__ void rgbmod_kernel(const float* __restrict__ w0,
                              const float* __restrict__ s,
                              float* __restrict__ rgbw, int Ci)
{
    int idx = blockIdx.x * blockDim.x + threadIdx.x;
    if (idx >= 4 * 3 * Ci) return;
    int ci = idx % Ci, c = (idx / Ci) % 3, b = idx / (3 * Ci);
    rgbw[idx] = rsqrtf((float)Ci) * w0[(size_t)c * Ci + ci] * s[(size_t)b * Ci + ci];
}

// const [1,512,4,4] NCHW f32 -> x f16 NHWC [4][16][512] tiled over batch
__global__ void const_init_kernel(const float* __restrict__ cst,
                                  _Float16* __restrict__ x)
{
    int idx = blockIdx.x * blockDim.x + threadIdx.x;
    if (idx >= 4 * 16 * 512) return;
    int c = idx % 512, p = (idx / 512) % 16;
    x[idx] = (_Float16)cst[c * 16 + p];
}

// --------------------------- WMMA implicit-GEMM conv -----------------------
// mode 0: 'same' 3x3 conv (pad 1), fused noise/bias/leakyReLU epilogue.
// mode 1: conv_transpose2d(stride 2) as dilated-valid conv (weights are
//         pre-flipped at pack time); raw output, epilogue comes after blur.
// Layouts: xin [B][Hi*Hi][Ci] f16, wgt [B][9][Co][Ci] f16, out [B][Ho*Ho][Co].
// One wave per 16x16 (Co x pixel) tile, 8 waves per block on distinct tiles.
// Invalid im2col columns read from the zero page -> branch-free inner loop.
__global__ __launch_bounds__(256)
void conv3x3_wmma(const _Float16* __restrict__ xin,
                  const _Float16* __restrict__ wgt,
                  _Float16* __restrict__ out,
                  const _Float16* __restrict__ zpage,
                  const float* __restrict__ noise,
                  const float* __restrict__ noise_w,
                  const float* __restrict__ act_b,
                  int Ci, int Co, int Hi, int Ho, int mode)
{
    const int lane = threadIdx.x & 31;
    const int wave = threadIdx.x >> 5;
    const int l16  = lane & 15;
    const int hid  = lane >> 4;
    const int HWo = Ho * Ho, HWi = Hi * Hi;
    const int numTiles = (HWo + 15) >> 4;
    const int tile = blockIdx.x * 8 + wave;
    const int co_base = blockIdx.y << 4;
    const int b = blockIdx.z;

    const int pix = tile * 16 + l16;
    const int py = pix / Ho, px = pix - py * Ho;
    const bool pv = (tile < numTiles) && (pix < HWo);

    const _Float16* xb = xin + (size_t)b * HWi * Ci;
    const _Float16* wb = wgt + (size_t)b * 9 * Co * Ci;

    const int ka0 = hid ? 16 : 0;    // A f16 layout: lanes 0-15 K 0-15,
    const int ka1 = hid ? 24 : 8;    //               lanes 16-31 K 16-31
    const int kb0 = hid ? 16 : 0;    // B f16 layout: K 0-15 / 16-31 per half

    v8f acc = {};
    for (int tap = 0; tap < 9; ++tap) {
        const int ky = tap / 3, kx = tap - ky * 3;
        int iy, ix;
        bool v = pv;
        if (mode == 0) {
            iy = py + ky - 1; ix = px + kx - 1;
            v = v && (iy >= 0) && (iy < Hi) && (ix >= 0) && (ix < Hi);
        } else {
            const int uy = py + ky, ux = px + kx;   // coords in padded dilated in
            v = v && (uy >= 2) && !(uy & 1) && (ux >= 2) && !(ux & 1);
            iy = (uy - 2) >> 1; ix = (ux - 2) >> 1;
            v = v && (iy < Hi) && (ix < Hi);
        }
        // zero-page redirect: unconditional vector loads, no EXEC round-trip
        const _Float16* xp =
            v ? (xb + ((size_t)iy * Hi + ix) * (size_t)Ci + kb0) : (zpage + kb0);
        const _Float16* wrow = wb + ((size_t)tap * Co + (co_base + l16)) * Ci;

        for (int c0 = 0; c0 < Ci; c0 += 32) {
            v16h A;
            {
                const v8h alo = *(const v8h*)(wrow + c0 + ka0);
                const v8h ahi = *(const v8h*)(wrow + c0 + ka1);
                #pragma unroll
                for (int e = 0; e < 8; ++e) { A[e] = alo[e]; A[e + 8] = ahi[e]; }
            }
            v16h Bm;
            {
                const v8h blo = *(const v8h*)(xp + c0);
                const v8h bhi = *(const v8h*)(xp + c0 + 8);
                #pragma unroll
                for (int e = 0; e < 8; ++e) { Bm[e] = blo[e]; Bm[e + 8] = bhi[e]; }
            }
            acc = __builtin_amdgcn_wmma_f32_16x16x32_f16(
                false, A, false, Bm, (short)0, acc, false, false);
        }
    }

    if (pv) {
        const int co0 = co_base + 8 * hid;   // D: lanes 0-15 M=r, 16-31 M=r+8
        v8h ov;
        if (mode == 0) {
            const float nz = noise_w[0] * noise[(size_t)b * HWo + pix];
            #pragma unroll
            for (int r = 0; r < 8; ++r)
                ov[r] = (_Float16)lrelu2(acc[r] + nz + act_b[co0 + r]);
        } else {
            #pragma unroll
            for (int r = 0; r < 8; ++r) ov[r] = (_Float16)acc[r];
        }
        *(v8h*)(out + ((size_t)b * HWo + pix) * Co + co0) = ov;
    }
}

// 4x4 binomial blur (k2d*4 => /16) over (2H+1)^2 -> (2H)^2, pad 1,
// fused with noise/bias/leakyReLU epilogue of the up-conv layer.
__global__ void blur_epi_kernel(const _Float16* __restrict__ y0,
                                _Float16* __restrict__ xo,
                                const float* __restrict__ noise,
                                const float* __restrict__ noise_w,
                                const float* __restrict__ act_b,
                                int C, int O, int Ho, int total)
{
    int idx = blockIdx.x * blockDim.x + threadIdx.x;
    if (idx >= total) return;
    const int c = idx % C;
    const int p = (idx / C) % (Ho * Ho);
    const int b = idx / (C * Ho * Ho);
    const int oy = p / Ho, ox = p - oy * Ho;
    const float K[4] = {1.f, 3.f, 3.f, 1.f};
    float sum = 0.f;
    for (int ky = 0; ky < 4; ++ky) {
        const int qy = oy + ky - 1;
        if (qy < 0 || qy >= O) continue;
        for (int kx = 0; kx < 4; ++kx) {
            const int qx = ox + kx - 1;
            if (qx < 0 || qx >= O) continue;
            sum += K[ky] * K[kx] *
                   (float)y0[(((size_t)b * O + qy) * O + qx) * C + c];
        }
    }
    sum *= (1.0f / 16.0f);
    sum += noise_w[0] * noise[(size_t)b * Ho * Ho + p] + act_b[c];
    xo[((size_t)b * Ho * Ho + p) * C + c] = (_Float16)lrelu2(sum);
}

// toRGB: skip[b][c][p] = bias[c] + sum_ci rgbw[b][c][ci] * x[b][p][ci]
__global__ void to_rgb_kernel(const _Float16* __restrict__ x,
                              const float* __restrict__ rgbw,
                              const float* __restrict__ bias,
                              float* __restrict__ skip, int Ci, int HW)
{
    int idx = blockIdx.x * blockDim.x + threadIdx.x;
    if (idx >= 4 * HW) return;
    const int p = idx % HW, b = idx / HW;
    const _Float16* xp = x + ((size_t)b * HW + p) * Ci;
    const float* r0 = rgbw + (size_t)(b * 3 + 0) * Ci;
    const float* r1 = rgbw + (size_t)(b * 3 + 1) * Ci;
    const float* r2 = rgbw + (size_t)(b * 3 + 2) * Ci;
    float a0 = bias[0], a1 = bias[1], a2 = bias[2];
    for (int c0 = 0; c0 < Ci; c0 += 8) {
        const v8h xv = *(const v8h*)(xp + c0);
        #pragma unroll
        for (int e = 0; e < 8; ++e) {
            const float f = (float)xv[e];
            a0 += r0[c0 + e] * f; a1 += r1[c0 + e] * f; a2 += r2[c0 + e] * f;
        }
    }
    skip[((size_t)b * 3 + 0) * HW + p] = a0;
    skip[((size_t)b * 3 + 1) * HW + p] = a1;
    skip[((size_t)b * 3 + 2) * HW + p] = a2;
}

// Skip path upfirdn2d(up=2, pad=(2,1)): zero-insert + pad 2 + 4x4 blur (/16),
// accumulated onto the new-resolution skip (NCHW f32).
__global__ void skip_up_add_kernel(const float* __restrict__ sin_,
                                   float* __restrict__ sout,
                                   int Hin, int Ho, int total)
{
    int idx = blockIdx.x * blockDim.x + threadIdx.x;
    if (idx >= total) return;
    const int p = idx % (Ho * Ho);
    const int bc = idx / (Ho * Ho);   // b*3+c
    const int oy = p / Ho, ox = p - oy * Ho;
    const float K[4] = {1.f, 3.f, 3.f, 1.f};
    float sum = 0.f;
    for (int ky = 0; ky < 4; ++ky) {
        const int u = oy + ky - 2;
        if (u < 0 || (u & 1)) continue;
        const int iy = u >> 1;
        if (iy >= Hin) continue;
        for (int kx = 0; kx < 4; ++kx) {
            const int w = ox + kx - 2;
            if (w < 0 || (w & 1)) continue;
            const int ix = w >> 1;
            if (ix >= Hin) continue;
            sum += K[ky] * K[kx] * sin_[(size_t)bc * Hin * Hin + iy * Hin + ix];
        }
    }
    sout[(size_t)bc * Ho * Ho + p] += sum * (1.0f / 16.0f);
}

// ---------------------------------------------------------------------------
extern "C" void kernel_launch(void* const* d_in, const int* in_sizes, int n_in,
                              void* d_out, int out_size, void* d_ws, size_t ws_size,
                              hipStream_t stream)
{
    (void)in_sizes; (void)n_in; (void)out_size; (void)ws_size;
    char* ws = (char*)d_ws;
    float*     wlat  = (float*)(ws);
    float*     sbuf  = (float*)(ws + (16 << 10));
    float*     rgbw  = (float*)(ws + (32 << 10));
    _Float16*  zpage = (_Float16*)(ws + (48 << 10));
    _Float16*  wgt   = (_Float16*)(ws + ((size_t)1   << 20));
    _Float16*  actA  = (_Float16*)(ws + ((size_t)24  << 20));
    _Float16*  actB  = (_Float16*)(ws + ((size_t)100 << 20));
    float*     skipA = (float*)(ws + ((size_t)176 << 20));
    float*     skipB = (float*)(ws + ((size_t)184 << 20));

    // Input order: zcode, mlp_w, mlp_b, const, 13x conv{w,mod_w,mod_b,noise_w,
    // act_b}, 7x rgb{w,mod_w,mod_b,bias}, 13x noise.
    const float* zcode = (const float*)d_in[0];
    const float* mlp_w = (const float*)d_in[1];
    const float* mlp_b = (const float*)d_in[2];
    const float* cst   = (const float*)d_in[3];
    auto convp  = [&](int j, int k) { return (const float*)d_in[4 + 5 * j + k]; };
    auto rgbp   = [&](int t, int k) { return (const float*)d_in[69 + 4 * t + k]; };
    auto noisep = [&](int j)        { return (const float*)d_in[97 + j]; };

    static const int CHS[7] = {512, 512, 512, 512, 512, 256, 128};
    static const int RES[7] = {4, 8, 16, 32, 64, 128, 256};

    zero_page_kernel<<<2, 256, 0, stream>>>(zpage);
    mapping_kernel<<<1, 512, 0, stream>>>(zcode, mlp_w, mlp_b, wlat);

    auto run_style = [&](const float* mw, const float* mb, int Ci) {
        int tot = 4 * Ci;
        style_kernel<<<(tot + 255) / 256, 256, 0, stream>>>(wlat, mw, mb, sbuf, Ci);
    };
    auto run_conv = [&](int j, int Ci, int Co, int Hi, int Ho, int mode,
                        const _Float16* xin, _Float16* xout) {
        run_style(convp(j, 1), convp(j, 2), Ci);
        modpack_kernel<<<dim3(Co, 4), 256, 0, stream>>>(convp(j, 0), sbuf, wgt,
                                                        Ci, Co, mode == 1);
        const int HWo = Ho * Ho;
        const int tiles = (HWo + 15) / 16;
        dim3 g((tiles + 7) / 8, Co / 16, 4);
        conv3x3_wmma<<<g, 256, 0, stream>>>(xin, wgt, xout, zpage, noisep(j),
                                            convp(j, 3), convp(j, 4),
                                            Ci, Co, Hi, Ho, mode);
    };
    auto run_rgb = [&](int t, int Ci, int HW, const _Float16* x, float* skip) {
        run_style(rgbp(t, 1), rgbp(t, 2), Ci);
        rgbmod_kernel<<<(4 * 3 * Ci + 255) / 256, 256, 0, stream>>>(
            rgbp(t, 0), sbuf, rgbw, Ci);
        to_rgb_kernel<<<(4 * HW + 255) / 256, 256, 0, stream>>>(
            x, rgbw, rgbp(t, 3), skip, Ci, HW);
    };

    _Float16* cur = actA;
    _Float16* alt = actB;

    const_init_kernel<<<(4 * 16 * 512 + 255) / 256, 256, 0, stream>>>(cst, cur);
    run_conv(0, 512, 512, 4, 4, 0, cur, alt);
    { _Float16* t = cur; cur = alt; alt = t; }
    run_rgb(0, 512, 16, cur, skipA);
    float* skip = skipA;

    for (int i = 0; i < 6; ++i) {
        const int Ci = CHS[i], Cm = CHS[i + 1];
        const int Hi = RES[i], Ho = RES[i + 1];
        const int O = 2 * Hi + 1;
        const int j1 = 1 + 2 * i, j2 = 2 + 2 * i;

        // transposed conv (raw, OxO) then blur + fused epilogue (HoxHo)
        run_conv(j1, Ci, Cm, Hi, O, 1, cur, alt);
        {
            int tot = 4 * Ho * Ho * Cm;
            blur_epi_kernel<<<(tot + 255) / 256, 256, 0, stream>>>(
                alt, cur, noisep(j1), convp(j1, 3), convp(j1, 4), Cm, O, Ho, tot);
        }
        // regular 'same' conv with fused epilogue
        run_conv(j2, Cm, Cm, Ho, Ho, 0, cur, alt);
        { _Float16* t = cur; cur = alt; alt = t; }

        float* snew = (i == 5) ? (float*)d_out
                               : ((skip == skipA) ? skipB : skipA);
        run_rgb(i + 1, Cm, Ho * Ho, cur, snew);
        {
            int tot = 4 * 3 * Ho * Ho;
            skip_up_add_kernel<<<(tot + 255) / 256, 256, 0, stream>>>(
                skip, snew, Hi, Ho, tot);
        }
        skip = snew;
    }
}
